// DWNBlock_1305670058052
// MI455X (gfx1250) — matrix operand order repaired and checked by reading, not compiled, not run on hardware
//
#include <hip/hip_runtime.h>
#include <hip/hip_bf16.h>
#include <math.h>

// ---------------- types ----------------
typedef __attribute__((ext_vector_type(16))) __bf16 v16bf;
typedef __attribute__((ext_vector_type(8)))  __bf16 v8bf;
typedef __attribute__((ext_vector_type(8)))  float  v8f;

union Frag { v16bf v; struct { v8bf lo, hi; } p; };

#define D_MODEL 128
#define D_STATE 128
#define HDIM    512
#define BATCH_  8
#define SEQ_    8192
#define NTOK    (BATCH_*SEQ_)
#define NCHUNK  64
#define CHUNKL  (SEQ_/NCHUNK)
#define SQRT2F  1.41421356237309504880f
#define SCALEF  1.0f

// epilogue modes
#define EP_F32      0   // out f32 = acc*scale
#define EP_BF16     1   // out bf16 = acc*scale
#define EP_RELU     2   // out bf16 = relu(acc*scale*avec[c])*bvec[c]
#define EP_RESID    3   // out f32 = acc*scale + resid[row*N+c]

// ---------------------------------------------------------------------------
// Generic WMMA GEMM: out[M x N] = epilogue( A[M x K](bf16) @ Wt[N x K]^T )
// Wt stored N-major (row n holds K contiguous bf16) so B fragments are
// two contiguous 16B loads. Block = 256 threads = 8 waves; block tile = 128 x 64;
// each wave computes a 16 x 64 strip (4 accumulators of v8f).
//
// K is always a multiple of 64 here (128/384/512), so the K loop advances by
// 64 with two STATICALLY-NAMED fragment stages (no runtime-indexed arrays ->
// no scratch spills): stage1's loads issue before stage0's WMMAs, stage0's
// reload issues before stage1's WMMAs. Loads overlap matrix math; the wait
// in front of each WMMA group only covers loads issued one stage earlier.
//
// Fragment layouts per CDNA5 ISA 7.12.2:
//   A 16x32 bf16 : lane L<16 -> row L, K = k0+[0..7] then k0+[16..23]
//                  lane L+16 -> row L, K = k0+[8..15] then k0+[24..31]
//   B 32x16 bf16 : lane L<16 -> col L, K = k0+[0..7] then k0+[16..23]
//                  lane L+16 -> col L, K = k0+[8..15] then k0+[24..31]
// ---------------------------------------------------------------------------
#define WMMA_BF16(ACC, AF, BF) \
  ACC = __builtin_amdgcn_wmma_f32_16x16x32_bf16( \
      false, (AF).v, false, (BF).v, (short)0, ACC, false, false)

__global__ __launch_bounds__(256)
void gemm_wmma(const __bf16* __restrict__ A, const __bf16* __restrict__ Wt,
               int K, int N, void* __restrict__ out, int mode, float scale,
               const float* __restrict__ avec, const float* __restrict__ bvec,
               const float* __restrict__ resid)
{
  const int lane = threadIdx.x & 31;
  const int wave = threadIdx.x >> 5;
  const int lr   = lane & 15;
  const int hi   = lane >> 4;
  const int m0   = blockIdx.x * 128 + wave * 16;
  const int n0   = blockIdx.y * 64;

  v8f acc0 = {}, acc1 = {}, acc2 = {}, acc3 = {};

  const __bf16* arow  = A  + (size_t)(m0 + lr) * K + hi * 8;
  const __bf16* brow0 = Wt + (size_t)(n0 +  0 + lr) * K + hi * 8;
  const __bf16* brow1 = Wt + (size_t)(n0 + 16 + lr) * K + hi * 8;
  const __bf16* brow2 = Wt + (size_t)(n0 + 32 + lr) * K + hi * 8;
  const __bf16* brow3 = Wt + (size_t)(n0 + 48 + lr) * K + hi * 8;

  auto ldA = [&](int k) {
    Frag f;
    f.p.lo = *(const v8bf*)(arow + k);
    f.p.hi = *(const v8bf*)(arow + k + 16);
    return f;
  };
  auto ldB = [&](const __bf16* br, int k) {
    Frag f;
    f.p.lo = *(const v8bf*)(br + k);
    f.p.hi = *(const v8bf*)(br + k + 16);
    return f;
  };

  // stage 0: K-step k0 ; stage 1: K-step k0+32  (all distinct named regs)
  Frag a0  = ldA(0);
  Frag b00 = ldB(brow0, 0), b01 = ldB(brow1, 0),
       b02 = ldB(brow2, 0), b03 = ldB(brow3, 0);

  for (int k0 = 0; k0 < K; k0 += 64) {
    const int ks1 = k0 + 32;   // always < K (K % 64 == 0)
    Frag a1  = ldA(ks1);
    Frag b10 = ldB(brow0, ks1), b11 = ldB(brow1, ks1),
         b12 = ldB(brow2, ks1), b13 = ldB(brow3, ks1);
    __builtin_prefetch(arow + k0 + 64, 0, 1);   // global_prefetch_b8

    WMMA_BF16(acc0, a0, b00);
    WMMA_BF16(acc1, a0, b01);
    WMMA_BF16(acc2, a0, b02);
    WMMA_BF16(acc3, a0, b03);

    const int ks2 = k0 + 64;
    if (ks2 < K) {
      a0  = ldA(ks2);
      b00 = ldB(brow0, ks2); b01 = ldB(brow1, ks2);
      b02 = ldB(brow2, ks2); b03 = ldB(brow3, ks2);
    }

    WMMA_BF16(acc0, a1, b10);
    WMMA_BF16(acc1, a1, b11);
    WMMA_BF16(acc2, a1, b12);
    WMMA_BF16(acc3, a1, b13);
  }

  // C/D layout: VGPR r: lanes 0-15 -> M=r, N=lane; lanes 16-31 -> M=8+r, N=lane-16
  v8f accs[4] = {acc0, acc1, acc2, acc3};
#pragma unroll
  for (int t = 0; t < 4; ++t) {
    int c = n0 + t * 16 + lr;
#pragma unroll
    for (int r = 0; r < 8; ++r) {
      int row = m0 + hi * 8 + r;
      float v = accs[t][r] * scale;
      size_t o = (size_t)row * N + c;
      if (mode == EP_F32) {
        ((float*)out)[o] = v;
      } else if (mode == EP_BF16) {
        ((__bf16*)out)[o] = (__bf16)v;
      } else if (mode == EP_RELU) {
        float u = v * avec[c];
        u = u > 0.f ? u : 0.f;
        ((__bf16*)out)[o] = (__bf16)(u * bvec[c]);
      } else { // EP_RESID
        ((float*)out)[o] = v + resid[o];
      }
    }
  }
}

// ---------------- Cayley preprocessing ----------------
// Frobenius norm of W (single block)
__global__ void sumsq_norm(const float* __restrict__ W, int n, float* outNorm) {
  __shared__ float red[256];
  float s = 0.f;
  for (int i = threadIdx.x; i < n; i += 256) { float w = W[i]; s += w * w; }
  red[threadIdx.x] = s; __syncthreads();
  for (int o = 128; o > 0; o >>= 1) {
    if (threadIdx.x < o) red[threadIdx.x] += red[threadIdx.x + o];
    __syncthreads();
  }
  if (threadIdx.x == 0) *outNorm = sqrtf(red[0]);
}

// Gaug = [I+A | I] (fout x 2*fout), ImA = I-A, where for Ws = s*W (s=alpha/norm):
// A[i][j] = Ws[j][i] - Ws[i][j] + sum_k Ws[i][fout+k]*Ws[j][fout+k]
__global__ void build_G(const float* __restrict__ W, int fout, int fin,
                        const float* alpha, const float* norm,
                        float* __restrict__ Gaug, float* __restrict__ ImA) {
  int i = blockIdx.x;
  float s = alpha[0] / norm[0];
  int kd = fin - fout;
  for (int j = threadIdx.x; j < fout; j += blockDim.x) {
    const float* wi = W + (size_t)i * fin + fout;
    const float* wj = W + (size_t)j * fin + fout;
    float dot = 0.f;
    for (int k = 0; k < kd; ++k) dot += wi[k] * wj[k];
    float a = s * (W[(size_t)j * fin + i] - W[(size_t)i * fin + j]) + s * s * dot;
    float id = (i == j) ? 1.f : 0.f;
    Gaug[(size_t)i * (2 * fout) + j]        = id + a;
    Gaug[(size_t)i * (2 * fout) + fout + j] = id;
    ImA [(size_t)i * fout + j]              = id - a;
  }
}

// In-place Gauss-Jordan on [G | I] -> [I | inv(G)]; I+A is well conditioned
// (identity + skew + PSD), so no pivoting needed. One block.
__global__ __launch_bounds__(512)
void gauss_jordan(float* __restrict__ G, int n) {
  const int tid = threadIdx.x;
  const int twon = 2 * n;
  for (int k = 0; k < n; ++k) {
    float piv = G[(size_t)k * twon + k];
    __syncthreads();
    for (int c = tid; c < twon; c += blockDim.x)
      G[(size_t)k * twon + c] /= piv;
    __syncthreads();
    if (tid < n && tid != k) {
      float f = G[(size_t)tid * twon + k];
      for (int c = 0; c < twon; ++c)
        G[(size_t)tid * twon + c] -= f * G[(size_t)k * twon + c];
    }
    __syncthreads();
  }
}

// P1^T (N=fout x K=fin-fout) in bf16:  T[n][k] = -2*s * sum_j W[j][fout+k]*M[j][n]
__global__ void make_T1(const float* __restrict__ W, int fout, int fin,
                        const float* alpha, const float* norm,
                        const float* __restrict__ Gaug, __bf16* __restrict__ T) {
  int n = blockIdx.x;
  int kd = fin - fout;
  float s = alpha[0] / norm[0];
  for (int k = threadIdx.x; k < kd; k += blockDim.x) {
    float acc = 0.f;
    for (int j = 0; j < fout; ++j)
      acc += W[(size_t)j * fin + fout + k] * Gaug[(size_t)j * (2 * fout) + fout + n];
    T[(size_t)n * kd + k] = (__bf16)(-2.f * s * acc);
  }
}

// P2^T (N=f x K=f) in bf16:  T[n][k] = sum_r ImA[r][k] * M[n][r]
__global__ void make_T2(const float* __restrict__ ImA, const float* __restrict__ Gaug,
                        int f, __bf16* __restrict__ T) {
  int n = blockIdx.x;
  for (int k = threadIdx.x; k < f; k += blockDim.x) {
    float acc = 0.f;
    for (int r = 0; r < f; ++r)
      acc += ImA[(size_t)r * f + k] * Gaug[(size_t)n * (2 * f) + f + r];
    T[(size_t)n * f + k] = (__bf16)acc;
  }
}

// ---------------- LRU prep ----------------
__global__ void lam_kernel(const float* nu, const float* th,
                           float* lam_re, float* lam_im) {
  int c = threadIdx.x;
  float la = expf(-expf(nu[c]));
  float ph = expf(th[c]);
  lam_re[c] = la * cosf(ph);
  lam_im[c] = la * sinf(ph);
}

// Wb (256 x 128) bf16: rows 0..127 = e^gamma*B_re, rows 128..255 = e^gamma*B_im
__global__ void wb_kernel(const float* gl, const float* Bre, const float* Bim,
                          __bf16* Wb) {
  int n = blockIdx.x, d = threadIdx.x;
  float g = expf(gl[n]);
  Wb[(size_t)n * D_MODEL + d]               = (__bf16)(g * Bre[n * D_MODEL + d]);
  Wb[(size_t)(n + D_STATE) * D_MODEL + d]   = (__bf16)(g * Bim[n * D_MODEL + d]);
}

// Wc (128 x 384) bf16: [C_re | -C_im | D]
__global__ void wc_kernel(const float* Cre, const float* Cim, const float* Dm,
                          __bf16* Wc) {
  int n = blockIdx.x, k = threadIdx.x;
  Wc[(size_t)n * 384 + k]       = (__bf16)Cre[n * D_STATE + k];
  Wc[(size_t)n * 384 + 128 + k] = (__bf16)(-Cim[n * D_STATE + k]);
  Wc[(size_t)n * 384 + 256 + k] = (__bf16)Dm[n * D_MODEL + k];
}

__global__ void psi_kernel(const float* psi, float* a, float* b) {
  int i = blockIdx.x * blockDim.x + threadIdx.x;
  if (i < HDIM) { a[i] = expf(-psi[i]) * SQRT2F; b[i] = expf(psi[i]); }
}

__global__ void tobf16(const float* __restrict__ x, __bf16* __restrict__ y, size_t n) {
  size_t i = blockIdx.x * (size_t)blockDim.x + threadIdx.x;
  if (i < n) y[i] = (__bf16)x[i];
}

// ---------------- chunked LRU scan (lambda constant per channel) ----------------
__global__ void scan_pass1(const float* __restrict__ Bu, const float* lre,
                           const float* lim, float* __restrict__ chend) {
  int b = blockIdx.x, ch = blockIdx.y, c = threadIdx.x;
  float lr = lre[c], li = lim[c], sr = 0.f, si = 0.f;
  const float* bu = Bu + ((size_t)b * SEQ_ + (size_t)ch * CHUNKL) * 256;
  for (int t = 0; t < CHUNKL; ++t) {
    float br = bu[(size_t)t * 256 + c], bi = bu[(size_t)t * 256 + 128 + c];
    float nr = lr * sr - li * si + br;
    float ni = lr * si + li * sr + bi;
    sr = nr; si = ni;
  }
  size_t o = ((size_t)(b * NCHUNK + ch)) * 256;
  chend[o + c] = sr; chend[o + 128 + c] = si;
}

__global__ void scan_pass2(const float* __restrict__ chend, const float* lre,
                           const float* lim, float* __restrict__ carry) {
  int b = blockIdx.x, c = threadIdx.x;
  float lr = lre[c], li = lim[c];
  // lam^CHUNKL
  float pr = 1.f, pi = 0.f;
  for (int t = 0; t < CHUNKL; ++t) {
    float nr = pr * lr - pi * li, ni = pr * li + pi * lr;
    pr = nr; pi = ni;
  }
  float cr = 0.f, ci = 0.f;
  size_t o0 = ((size_t)(b * NCHUNK)) * 256;
  carry[o0 + c] = 0.f; carry[o0 + 128 + c] = 0.f;
  for (int k = 1; k < NCHUNK; ++k) {
    size_t prev = ((size_t)(b * NCHUNK + k - 1)) * 256;
    float er = chend[prev + c], ei = chend[prev + 128 + c];
    float nr = pr * cr - pi * ci + er;
    float ni = pr * ci + pi * cr + ei;
    cr = nr; ci = ni;
    size_t o = ((size_t)(b * NCHUNK + k)) * 256;
    carry[o + c] = cr; carry[o + 128 + c] = ci;
  }
}

// pass3: rescan with carry-in, emit Sx = [s_re | s_im | x] bf16 (K=384) per token
__global__ void scan_pass3(const float* __restrict__ Bu, const float* __restrict__ x,
                           const float* lre, const float* lim,
                           const float* __restrict__ carry, __bf16* __restrict__ Sx) {
  int b = blockIdx.x, ch = blockIdx.y, c = threadIdx.x;
  float lr = lre[c], li = lim[c];
  size_t ci0 = ((size_t)(b * NCHUNK + ch)) * 256;
  float sr = carry[ci0 + c], si = carry[ci0 + 128 + c];
  size_t tok0 = (size_t)b * SEQ_ + (size_t)ch * CHUNKL;
  for (int t = 0; t < CHUNKL; ++t) {
    size_t tok = tok0 + t;
    float br = Bu[tok * 256 + c], bi = Bu[tok * 256 + 128 + c];
    float nr = lr * sr - li * si + br;
    float ni = lr * si + li * sr + bi;
    sr = nr; si = ni;
    Sx[tok * 384 + c]         = (__bf16)sr;
    Sx[tok * 384 + 128 + c]   = (__bf16)si;
    Sx[tok * 384 + 256 + c]   = (__bf16)x[tok * 128 + c];
  }
}

// ---------------- host orchestration ----------------
extern "C" void kernel_launch(void* const* d_in, const int* in_sizes, int n_in,
                              void* d_out, int out_size, void* d_ws, size_t ws_size,
                              hipStream_t stream) {
  const float* x     = (const float*)d_in[0];
  const float* nu    = (const float*)d_in[1];
  const float* th    = (const float*)d_in[2];
  const float* gl    = (const float*)d_in[3];
  const float* Bre   = (const float*)d_in[4];
  const float* Bim   = (const float*)d_in[5];
  const float* Cre   = (const float*)d_in[6];
  const float* Cim   = (const float*)d_in[7];
  const float* Dm    = (const float*)d_in[8];
  const float* W1    = (const float*)d_in[9];
  const float* al1   = (const float*)d_in[10];
  const float* psi1  = (const float*)d_in[11];
  const float* W2    = (const float*)d_in[12];
  const float* al2   = (const float*)d_in[13];
  const float* psi2  = (const float*)d_in[14];
  const float* W3    = (const float*)d_in[15];
  const float* al3   = (const float*)d_in[16];
  const float* psi3  = (const float*)d_in[17];
  const float* Wl    = (const float*)d_in[18];
  const float* all_  = (const float*)d_in[19];

  char* p = (char*)d_ws;
  auto alloc = [&](size_t bytes) -> void* {
    void* r = (void*)p;
    p += (bytes + 255) & ~(size_t)255;
    return r;
  };

  float*  Bu    = (float*) alloc((size_t)NTOK * 256 * 4);      // 64 MB
  __bf16* Sx    = (__bf16*)alloc((size_t)NTOK * 384 * 2);      // 48 MB
  __bf16* zb    = (__bf16*)alloc((size_t)NTOK * 128 * 2);      // 16 MB
  __bf16* actA  = (__bf16*)alloc((size_t)NTOK * 512 * 2);      // 64 MB
  __bf16* actB  = (__bf16*)alloc((size_t)NTOK * 512 * 2);      // 64 MB
  __bf16* xb    = (__bf16*)alloc((size_t)NTOK * 128 * 2);      // 16 MB
  float*  chend = (float*) alloc((size_t)BATCH_ * NCHUNK * 256 * 4);
  float*  carry = (float*) alloc((size_t)BATCH_ * NCHUNK * 256 * 4);
  float*  lamre = (float*) alloc(128 * 4);
  float*  lamim = (float*) alloc(128 * 4);
  __bf16* Wb    = (__bf16*)alloc(256 * 128 * 2);
  __bf16* Wc    = (__bf16*)alloc(128 * 384 * 2);
  __bf16* T1_1  = (__bf16*)alloc(512 * 128 * 2);
  __bf16* T2_1  = (__bf16*)alloc(512 * 512 * 2);
  __bf16* T1_2  = (__bf16*)alloc(512 * 512 * 2);
  __bf16* T2_2  = (__bf16*)alloc(512 * 512 * 2);
  __bf16* T1_3  = (__bf16*)alloc(512 * 512 * 2);
  __bf16* T2_3  = (__bf16*)alloc(512 * 512 * 2);
  __bf16* Tl    = (__bf16*)alloc(128 * 512 * 2);
  float*  Gaug  = (float*) alloc((size_t)512 * 1024 * 4);      // reused per weight
  float*  ImA   = (float*) alloc((size_t)512 * 512 * 4);       // reused per weight
  float*  norms = (float*) alloc(64);
  float*  av    = (float*) alloc(3 * 512 * 4);
  float*  bv    = (float*) alloc(3 * 512 * 4);
  (void)ws_size; (void)in_sizes; (void)n_in; (void)out_size;

  // ---- Cayley prep (stream-serialized; Gaug/ImA reused) ----
  sumsq_norm<<<1, 256, 0, stream>>>(W1, 512 * 640, norms + 0);
  build_G<<<512, 256, 0, stream>>>(W1, 512, 640, al1, norms + 0, Gaug, ImA);
  gauss_jordan<<<1, 512, 0, stream>>>(Gaug, 512);
  make_T1<<<512, 128, 0, stream>>>(W1, 512, 640, al1, norms + 0, Gaug, T1_1);
  make_T2<<<512, 256, 0, stream>>>(ImA, Gaug, 512, T2_1);

  sumsq_norm<<<1, 256, 0, stream>>>(W2, 512 * 1024, norms + 1);
  build_G<<<512, 256, 0, stream>>>(W2, 512, 1024, al2, norms + 1, Gaug, ImA);
  gauss_jordan<<<1, 512, 0, stream>>>(Gaug, 512);
  make_T1<<<512, 256, 0, stream>>>(W2, 512, 1024, al2, norms + 1, Gaug, T1_2);
  make_T2<<<512, 256, 0, stream>>>(ImA, Gaug, 512, T2_2);

  sumsq_norm<<<1, 256, 0, stream>>>(W3, 512 * 1024, norms + 2);
  build_G<<<512, 256, 0, stream>>>(W3, 512, 1024, al3, norms + 2, Gaug, ImA);
  gauss_jordan<<<1, 512, 0, stream>>>(Gaug, 512);
  make_T1<<<512, 256, 0, stream>>>(W3, 512, 1024, al3, norms + 2, Gaug, T1_3);
  make_T2<<<512, 256, 0, stream>>>(ImA, Gaug, 512, T2_3);

  sumsq_norm<<<1, 256, 0, stream>>>(Wl, 128 * 640, norms + 3);
  build_G<<<128, 256, 0, stream>>>(Wl, 128, 640, all_, norms + 3, Gaug, ImA);
  gauss_jordan<<<1, 512, 0, stream>>>(Gaug, 128);
  make_T1<<<128, 256, 0, stream>>>(Wl, 128, 640, all_, norms + 3, Gaug, Tl);

  // ---- LRU prep ----
  lam_kernel<<<1, 128, 0, stream>>>(nu, th, lamre, lamim);
  wb_kernel<<<128, 128, 0, stream>>>(gl, Bre, Bim, Wb);
  wc_kernel<<<128, 128, 0, stream>>>(Cre, Cim, Dm, Wc);
  psi_kernel<<<2, 256, 0, stream>>>(psi1, av + 0,    bv + 0);
  psi_kernel<<<2, 256, 0, stream>>>(psi2, av + 512,  bv + 512);
  psi_kernel<<<2, 256, 0, stream>>>(psi3, av + 1024, bv + 1024);
  tobf16<<<(NTOK * 128 + 255) / 256, 256, 0, stream>>>(x, xb, (size_t)NTOK * 128);

  const int MB = NTOK / 128;  // 512 block rows

  // Bu = x @ [Beff_re; Beff_im]^T  (K=128, N=256) -> fp32 for the scan
  gemm_wmma<<<dim3(MB, 256 / 64), 256, 0, stream>>>(
      xb, Wb, 128, 256, Bu, EP_F32, 1.0f, nullptr, nullptr, nullptr);

  // chunked complex scan over SEQ
  scan_pass1<<<dim3(BATCH_, NCHUNK), 128, 0, stream>>>(Bu, lamre, lamim, chend);
  scan_pass2<<<BATCH_, 128, 0, stream>>>(chend, lamre, lamim, carry);
  scan_pass3<<<dim3(BATCH_, NCHUNK), 128, 0, stream>>>(Bu, x, lamre, lamim, carry, Sx);

  // z = SCALE * ( s_re@C_re^T - s_im@C_im^T + x@D^T )  (K=384, N=128)
  gemm_wmma<<<dim3(MB, 128 / 64), 256, 0, stream>>>(
      Sx, Wc, 384, 128, zb, EP_BF16, SCALEF, nullptr, nullptr, nullptr);

  // sandwich 1
  gemm_wmma<<<dim3(MB, 512 / 64), 256, 0, stream>>>(
      zb, T1_1, 128, 512, actA, EP_RELU, SCALEF, av + 0, bv + 0, nullptr);
  gemm_wmma<<<dim3(MB, 512 / 64), 256, 0, stream>>>(
      actA, T2_1, 512, 512, actB, EP_BF16, SQRT2F, nullptr, nullptr, nullptr);
  // sandwich 2
  gemm_wmma<<<dim3(MB, 512 / 64), 256, 0, stream>>>(
      actB, T1_2, 512, 512, actA, EP_RELU, SCALEF, av + 512, bv + 512, nullptr);
  gemm_wmma<<<dim3(MB, 512 / 64), 256, 0, stream>>>(
      actA, T2_2, 512, 512, actB, EP_BF16, SQRT2F, nullptr, nullptr, nullptr);
  // sandwich 3
  gemm_wmma<<<dim3(MB, 512 / 64), 256, 0, stream>>>(
      actB, T1_3, 512, 512, actA, EP_RELU, SCALEF, av + 1024, bv + 1024, nullptr);
  gemm_wmma<<<dim3(MB, 512 / 64), 256, 0, stream>>>(
      actA, T2_3, 512, 512, actB, EP_BF16, SQRT2F, nullptr, nullptr, nullptr);
  // final linear + residual (K=512, N=128), fp32 out
  gemm_wmma<<<dim3(MB, 128 / 64), 256, 0, stream>>>(
      actB, Tl, 512, 128, d_out, EP_RESID, SCALEF, nullptr, nullptr, x);
}
